// SAGE_9612136808452
// MI455X (gfx1250) — compile-verified, matrix operand order checked by reading, not run on hardware
//
#include <hip/hip_runtime.h>

typedef __attribute__((ext_vector_type(16))) _Float16 v16h;
typedef __attribute__((ext_vector_type(8)))  _Float16 v8h;
typedef __attribute__((ext_vector_type(8)))  float    v8f;

#define N_NODES 50000
#define N_EDGES 800000
#define D 128
#define MTILES (N_NODES / 16)     // 3125, exact
#define NTILES (D / 16)           // 8
#define KSTEPS (D / 32)           // 4
#define PACKW_ELEMS (NTILES * KSTEPS * 32 * 16)   // 16384 f16 per matrix

// ---------------- zero workspace ----------------
__global__ void SAGE_zero_kernel(float* __restrict__ p, int n) {
    int i = blockIdx.x * blockDim.x + threadIdx.x;
    int stride = gridDim.x * blockDim.x;
    for (; i < n; i += stride) p[i] = 0.0f;
}

// ---------------- f32 -> f16 row-major copy ----------------
__global__ void SAGE_cvt_kernel(const float* __restrict__ src,
                                _Float16* __restrict__ dst, int n) {
    int i = blockIdx.x * blockDim.x + threadIdx.x;
    int stride = gridDim.x * blockDim.x;
    for (; i < n; i += stride) dst[i] = (_Float16)src[i];
}

// ---------------- pack a [K=128, N=128] f32 weight matrix into -----------
// WMMA B-fragment order, f16. Output index:
//   P[(((ntile*KSTEPS + kk)*32 + lane)*16 + e]
// so each lane's 16-element fragment is a contiguous 32B run, and a wave's
// fragment is one coalesced 1KB block.
__global__ void SAGE_packW_kernel(const float* __restrict__ W,
                                  _Float16* __restrict__ P) {
    int idx = blockIdx.x * blockDim.x + threadIdx.x;
    if (idx >= PACKW_ELEMS) return;
    int e     = idx & 15;
    int lane  = (idx >> 4) & 31;
    int kk    = (idx >> 9) & 3;
    int ntile = idx >> 11;
    int half  = lane >> 4;
    int rc    = lane & 15;
    int col   = ntile * 16 + rc;
    int ka    = kk * 32 + half * 8;
    int k     = (e < 8) ? (ka + e) : (ka + 16 + (e - 8));
    P[idx] = (_Float16)W[k * D + col];
}

// ---------------- edge scatter: one wave32 per edge ----------------
// lane l handles float4 chunk l. Gather from h[src] is a coalesced 512B read
// (node matrix 25.6MB << 192MB L2); accumulate with global_atomic_add_f32.
__global__ __launch_bounds__(256)
void SAGE_scatter_kernel(const float* __restrict__ h,
                         const long long* __restrict__ src,
                         const long long* __restrict__ dst,
                         float* __restrict__ summed,
                         float* __restrict__ cnt) {
    int e = blockIdx.x * (blockDim.x >> 5) + (threadIdx.x >> 5);
    int lane = threadIdx.x & 31;
    if (e >= N_EDGES) return;
    int s = (int)src[e];
    int d = (int)dst[e];
    const float4* hv = (const float4*)(h + (size_t)s * D);
    float4 v = hv[lane];
    float* out = summed + (size_t)d * D + lane * 4;
    atomicAdd(out + 0, v.x);
    atomicAdd(out + 1, v.y);
    atomicAdd(out + 2, v.z);
    atomicAdd(out + 3, v.w);
    if (lane == 0) atomicAdd(cnt + d, 1.0f);
}

// ---------------- mean-normalize -> f16 (GEMM A operand) ----------------
// summed f32 is consumed only here; emit f16 directly, skip f32 writeback.
__global__ void SAGE_agg_kernel(const float* __restrict__ summed,
                                const float* __restrict__ cnt,
                                _Float16* __restrict__ aggh) {
    int i = blockIdx.x * blockDim.x + threadIdx.x;
    int total = N_NODES * D;
    int stride = gridDim.x * blockDim.x;
    for (; i < total; i += stride) {
        float c = cnt[i >> 7];               // D == 128
        aggh[i] = (_Float16)(summed[i] / fmaxf(c, 1.0f));
    }
}

// ---------------- fused SAGE GEMM: out = agg@Wl + self@Wr + b (opt ReLU) ----
// Block = 256 threads = 8 waves; wave w -> ntile w, block -> mtile. Per K-step
// of 32: 2x b128 A loads + 2x b128 packed-B loads + 1 v_wmma_f32_16x16x32_f16.
// Two passes (agg*Wl, self*Wr) share the f32 accumulator.
__global__ __launch_bounds__(256)
void SAGE_gemm_kernel(const _Float16* __restrict__ Aagg,
                      const _Float16* __restrict__ Aself,
                      const _Float16* __restrict__ PWl,  // packed fragments
                      const _Float16* __restrict__ PWr,
                      const float* __restrict__ bias,
                      float* __restrict__ out_f32,
                      _Float16* __restrict__ out_f16,    // may be null
                      int relu) {
    const int lane  = threadIdx.x & 31;
    const int wave  = threadIdx.x >> 5;      // ntile 0..7
    const int mtile = blockIdx.x;            // 0..3124
    const int half  = lane >> 4;
    const int rc    = lane & 15;
    const int row   = mtile * 16 + rc;
    const int col   = wave * 16 + rc;

    v8f acc = {};

    #pragma unroll
    for (int pass = 0; pass < 2; ++pass) {
        const _Float16* __restrict__ A = pass ? Aself : Aagg;
        const _Float16* __restrict__ P = pass ? PWr : PWl;
        const _Float16* __restrict__ arow = A + (size_t)row * D;
        // base of this wave's packed fragments: fragment kk at +kk*512
        const _Float16* __restrict__ pbase =
            P + ((size_t)wave * KSTEPS * 32 + lane) * 16;

        #pragma unroll
        for (int kk = 0; kk < KSTEPS; ++kk) {
            const int ka = kk * 32 + half * 8;
            // A fragment: ISA 16-bit A 16x32 layout; two contiguous 8xf16 runs
            v8h lo = *(const v8h*)(arow + ka);
            v8h hi = *(const v8h*)(arow + ka + 16);
            v16h af, bf;
            #pragma unroll
            for (int e = 0; e < 8; ++e) { af[e] = lo[e]; af[e + 8] = hi[e]; }
            // B fragment: one 32B contiguous per-lane load from packed weights
            bf = *(const v16h*)(pbase + (size_t)kk * 32 * 16);
            acc = __builtin_amdgcn_wmma_f32_16x16x32_f16(
                false, af, false, bf, (short)0, acc, false, false);
        }
    }

    // Epilogue: D layout — VGPR v, lane L: m = v + (L>>4)*8, n = L&15.
    const float bv = bias[col];
    #pragma unroll
    for (int v = 0; v < 8; ++v) {
        const int m = mtile * 16 + half * 8 + v;
        float val = acc[v] + bv;
        if (relu) val = fmaxf(val, 0.0f);
        out_f32[(size_t)m * D + col] = val;
        if (out_f16) out_f16[(size_t)m * D + col] = (_Float16)val;
    }
}

extern "C" void kernel_launch(void* const* d_in, const int* in_sizes, int n_in,
                              void* d_out, int out_size, void* d_ws, size_t ws_size,
                              hipStream_t stream) {
    const float*     x    = (const float*)d_in[0];
    const long long* ei   = (const long long*)d_in[1];
    const float*     Wl1  = (const float*)d_in[2];
    const float*     Wr1  = (const float*)d_in[3];
    const float*     b1   = (const float*)d_in[4];
    const float*     Wl2  = (const float*)d_in[5];
    const float*     Wr2  = (const float*)d_in[6];
    const float*     b2   = (const float*)d_in[7];
    float*           out  = (float*)d_out;

    const long long* src = ei;            // edge_index[0]
    const long long* dst = ei + N_EDGES;  // edge_index[1]

    const size_t ND = (size_t)N_NODES * D;
    float*    summed = (float*)d_ws;                    // ND f32
    float*    cnt    = summed + ND;                     // N f32
    float*    h      = cnt + N_NODES;                   // ND f32
    _Float16* xh     = (_Float16*)(h + ND);             // ND f16
    _Float16* aggh   = xh + ND;                         // ND f16
    _Float16* hh     = aggh + ND;                       // ND f16
    _Float16* pw1l   = hh + ND;                         // 16384 f16 each
    _Float16* pw1r   = pw1l + PACKW_ELEMS;
    _Float16* pw2l   = pw1r + PACKW_ELEMS;
    _Float16* pw2r   = pw2l + PACKW_ELEMS;

    const int nZero = N_NODES * D + N_NODES;
    const int zeroBlocks = (nZero + 255) / 256;
    const int scatBlocks = N_EDGES / 8;                 // 8 edges / 256-thr block
    const int elemBlocks = (int)((ND + 255) / 256);
    const int packBlocks = (PACKW_ELEMS + 255) / 256;

    // ---- one-shot prep (re-run every call: deterministic) ----
    SAGE_packW_kernel<<<packBlocks, 256, 0, stream>>>(Wl1, pw1l);
    SAGE_packW_kernel<<<packBlocks, 256, 0, stream>>>(Wr1, pw1r);
    SAGE_packW_kernel<<<packBlocks, 256, 0, stream>>>(Wl2, pw2l);
    SAGE_packW_kernel<<<packBlocks, 256, 0, stream>>>(Wr2, pw2r);
    SAGE_cvt_kernel<<<elemBlocks, 256, 0, stream>>>(x, xh, (int)ND);

    // ---- Layer 1 ----
    SAGE_zero_kernel<<<zeroBlocks, 256, 0, stream>>>(summed, nZero);
    SAGE_scatter_kernel<<<scatBlocks, 256, 0, stream>>>(x, src, dst, summed, cnt);
    SAGE_agg_kernel<<<elemBlocks, 256, 0, stream>>>(summed, cnt, aggh);
    SAGE_gemm_kernel<<<MTILES, 256, 0, stream>>>(aggh, xh, pw1l, pw1r, b1,
                                                 h, hh, 1);

    // ---- Layer 2 ----
    SAGE_zero_kernel<<<zeroBlocks, 256, 0, stream>>>(summed, nZero);
    SAGE_scatter_kernel<<<scatBlocks, 256, 0, stream>>>(h, src, dst, summed, cnt);
    SAGE_agg_kernel<<<elemBlocks, 256, 0, stream>>>(summed, cnt, aggh);
    SAGE_gemm_kernel<<<MTILES, 256, 0, stream>>>(aggh, hh, pw2l, pw2r, b2,
                                                 out, (_Float16*)nullptr, 0);
}
